// SparseConstraint_15350213116416
// MI455X (gfx1250) — matrix operand order repaired and checked by reading, not compiled, not run on hardware
//
#include <hip/hip_runtime.h>
#include <hip/hip_bf16.h>
#include <math.h>

typedef __attribute__((ext_vector_type(16))) _Float16 v16h;
typedef __attribute__((ext_vector_type(8)))  _Float16 v8h;
typedef __attribute__((ext_vector_type(8)))  float    v8f;

#define WAVES 16   // waves per workgroup = patches per workgroup

// CDNA5 wave32 WMMA f16 fragment layouts (CDNA5 ISA):
//  A 16x32 f16 : lane L row M=L&15; element e -> K = e + 8*(e>>3) + 8*(L>>4)
//                => per lane two contiguous K-chunks: [8hi..8hi+7], [8hi+16..8hi+23]
//  B 32x16 f16 : lane L col N=L&15; element e -> K = e + 8*(e>>3) + 8*(L>>4) (same as A)
//                => same two K-chunks as A: [8hi..8hi+7], [8hi+16..8hi+23]
//  C/D 16x16 f32: lane L col N=L&15; vgpr r -> M = r + 8*(L>>4)

__global__ __launch_bounds__(32 * WAVES) void fused_patch_kernel(
    const float* __restrict__ x,
    const float* __restrict__ text_embs,   // [4,768]
    const float* __restrict__ center,      // [2]
    const float* __restrict__ trw,         // [768,32]
    const float* __restrict__ trb,         // [32]
    const float* __restrict__ c1w, const float* __restrict__ c1b,   // [16,1,3,3],[16]
    const float* __restrict__ c2w, const float* __restrict__ c2b,   // [32,16,3,3],[32]
    const float* __restrict__ m1w, const float* __restrict__ m1b,   // [65,64],[64]
    const float* __restrict__ m2w, const float* __restrict__ m2b,   // [64,32],[32]
    const float* __restrict__ m3w, const float* __restrict__ m3b,   // [32,1],[1]
    float* __restrict__ sparsity_out)      // [B*16384]
{
  // Zero-padded f16 operand staging (all WMMA inputs assembled from these)
  __shared__ _Float16 s_c1bt[16 * 32];          //  1 KB  B^T conv1 [N=16][K=32]
  __shared__ _Float16 s_c2bt[32 * 160];         // 10 KB  B^T conv2 [N=32][K=160]
  __shared__ _Float16 s_patchp[WAVES * 100];    //  3.1KB padded patch [10][10]
  __shared__ _Float16 s_ph[WAVES * 2 * 16 * 16];// 16 KB  half-pooled [hi][cell][ch]
  __shared__ _Float16 s_poolp[WAVES * 576];     // 18 KB  padded pooled [6][6][16]
  __shared__ float    s_red[WAVES][64];         //  4 KB  avg-pool partials
  __shared__ _Float16 s_comb[16][96];           //  3 KB  MLP input tile [M][K]
  __shared__ _Float16 s_h1[16][64];             //  2 KB
  __shared__ _Float16 s_h2[16][32];             //  1 KB
  __shared__ float    s_textf[32];

  const int tid  = threadIdx.x;
  const int w    = tid >> 5;    // wave id (= patch slot)
  const int lane = tid & 31;
  const int lo   = lane & 15;
  const int hi   = lane >> 4;

  const int p  = blockIdx.x * WAVES + w;   // global patch id
  const int b  = p >> 14;                  // 16384 patches per batch
  const int pi = p & 16383;
  const int py = pi >> 7;
  const int px = pi & 127;

  // ================= phase 0: staging =================
  // patch interior (2 floats/lane -> f16), padded coords [+1][+1]
  {
    const float* xb = x + ((size_t)b << 20) + (size_t)(py * 8) * 1024 + px * 8;
    int i0 = lane * 2;
    int r = i0 >> 3, c = i0 & 7;
    float2 v = *(const float2*)(xb + r * 1024 + c);
    s_patchp[w * 100 + (r + 1) * 10 + (c + 1)]     = (_Float16)v.x;
    s_patchp[w * 100 + (r + 1) * 10 + (c + 1) + 1] = (_Float16)v.y;
  }
  // patch border zeros
  for (int i = lane; i < 100; i += 32) {
    int r = i / 10, c = i % 10;
    if (r == 0 || r == 9 || c == 0 || c == 9) s_patchp[w * 100 + i] = (_Float16)0.0f;
  }
  // pooled border zeros ([6][6][16], border cells)
  for (int i = lane; i < 576; i += 32) {
    int cell = i >> 4, r = cell / 6, c = cell % 6;
    if (r == 0 || r == 5 || c == 0 || c == 5) s_poolp[w * 576 + i] = (_Float16)0.0f;
  }
  // weight staging (once per block, all 512 threads)
  for (int i = tid; i < 16 * 32; i += 32 * WAVES) {        // c1BT[n][k]
    int n = i >> 5, k = i & 31;
    s_c1bt[i] = (k < 9) ? (_Float16)c1w[n * 9 + k] : (_Float16)0.0f;
  }
  for (int i = tid; i < 32 * 160; i += 32 * WAVES) {       // c2BT[n][k]
    int n = i / 160, k = i - n * 160;
    s_c2bt[i] = (k < 144) ? (_Float16)c2w[n * 144 + k] : (_Float16)0.0f;
  }
  if (w == 1) {   // text_f[b] = text_embs[b] @ tr_w + tr_b (all patches share b)
    float s = trb[lane];
    const float* te = text_embs + b * 768;
    for (int k = 0; k < 768; k++) s += te[k] * trw[k * 32 + lane];
    s_textf[lane] = s;
  }
  __syncthreads();

  // ====== phase 1: conv1 (1->16, 3x3 SAME on 8x8) + fused horizontal pool ======
  {
    v16h bf;
    ((v8h*)&bf)[0] = *reinterpret_cast<const v8h*>(&s_c1bt[lo * 32 + 8 * hi]);
    ((v8h*)&bf)[1] = *reinterpret_cast<const v8h*>(&s_c1bt[lo * 32 + 16 + 8 * hi]);
    float bias = c1b[lo];
#pragma unroll
    for (int t = 0; t < 4; t++) {              // M-tile t = pooled row band y
      int m = 16 * t + lo;
      int pbase = w * 100 + ((m >> 3) + 1) * 10 + (m & 7) + 1;
      int zabs  = w * 100;                     // patchp[w][0][0] == 0
      v16h af;
#pragma unroll
      for (int e = 0; e < 16; e++) {
        if (e >= 8) { af[e] = (_Float16)0.0f; continue; }   // K>=16|24: zero pad
        int off_lo = (e / 3 - 1) * 10 + (e % 3 - 1);        // tap e (lo half)
        int addr = (e == 0) ? (pbase + (hi ? 11 : off_lo))  // hi half: tap 8
                            : (hi ? zabs : pbase + off_lo); // hi half e>0: zero
        af[e] = s_patchp[addr];
      }
      v8f acc = {};
      acc = __builtin_amdgcn_wmma_f32_16x16x32_f16(false, af, false, bf,
                                                   (short)0, acc, false, false);
      // lane's 8 accs = image row (2t+hi), cols 0..7 -> horizontal pool pairs
#pragma unroll
      for (int xp = 0; xp < 4; xp++) {
        float hm = fmaxf(fmaxf(acc[2 * xp], acc[2 * xp + 1]) + bias, 0.0f);
        s_ph[((w * 2 + hi) * 16 + t * 4 + xp) * 16 + lo] = (_Float16)hm;
      }
    }
  }
  __syncthreads();

  // ====== phase 2: vertical pool combine -> padded pooled [6][6][16] ======
#pragma unroll
  for (int i = 0; i < 8; i++) {
    int idx = lane * 8 + i;                    // over [16 cells][16 ch]
    int cell = idx >> 4, ch = idx & 15;
    int yy = cell >> 2, xx = cell & 3;
    _Float16 a = s_ph[((w * 2 + 0) * 16 + cell) * 16 + ch];
    _Float16 c = s_ph[((w * 2 + 1) * 16 + cell) * 16 + ch];
    s_poolp[w * 576 + ((yy + 1) * 6 + (xx + 1)) * 16 + ch] = (a > c) ? a : c;
  }
  __syncthreads();

  // ====== phase 3: conv2 (16->32, 3x3 SAME on 4x4), K=144 pad 160 ======
  {
    v8f acc0 = {}, acc1 = {};
    int pbase = w * 576 + (((lo >> 2) + 1) * 6 + (lo & 3) + 1) * 16;  // pixel M=lo
#pragma unroll
    for (int ks = 0; ks < 5; ks++) {
      v16h af;
#pragma unroll
      for (int e = 0; e < 16; e++) {
        int k0 = 32 * ks + e + 8 * (e >> 3);   // hi=0 K index (compile-time)
        if (ks == 4 && e >= 8) { af[e] = (_Float16)0.0f; continue; }  // K>=144
        int k1 = k0 + 8;                        // hi=1 K index
        int t0 = k0 % 9, t1 = k1 % 9;
        int off0 = ((t0 / 3 - 1) * 6 + (t0 % 3 - 1)) * 16 + (k0 / 9);
        int off1 = ((t1 / 3 - 1) * 6 + (t1 % 3 - 1)) * 16 + (k1 / 9);
        af[e] = s_poolp[pbase + (hi ? off1 : off0)];
      }
#pragma unroll
      for (int nt = 0; nt < 2; nt++) {
        v16h bf;
        ((v8h*)&bf)[0] = *reinterpret_cast<const v8h*>(
            &s_c2bt[(nt * 16 + lo) * 160 + 32 * ks + 8 * hi]);
        ((v8h*)&bf)[1] = *reinterpret_cast<const v8h*>(
            &s_c2bt[(nt * 16 + lo) * 160 + 32 * ks + 16 + 8 * hi]);
        if (nt == 0)
          acc0 = __builtin_amdgcn_wmma_f32_16x16x32_f16(false, af, false, bf,
                                                        (short)0, acc0, false, false);
        else
          acc1 = __builtin_amdgcn_wmma_f32_16x16x32_f16(false, af, false, bf,
                                                        (short)0, acc1, false, false);
      }
    }
    // relu + partial row-sums for global average pool
    float s0 = 0.0f, s1 = 0.0f;
    float b0 = c2b[lo], b1 = c2b[16 + lo];
#pragma unroll
    for (int r = 0; r < 8; r++) {
      s0 += fmaxf(acc0[r] + b0, 0.0f);
      s1 += fmaxf(acc1[r] + b1, 0.0f);
    }
    s_red[w][hi * 16 + lo]      = s0;
    s_red[w][32 + hi * 16 + lo] = s1;
  }
  __syncthreads();

  // ====== phase 4: comb row [vis(32) | nd(1) | text(32) | pad -> 96] ======
  {
    int nt = lane >> 4, n = lane & 15;
    float vis = (s_red[w][nt * 32 + n] + s_red[w][nt * 32 + 16 + n]) * (1.0f / 16.0f);
    s_comb[w][lane] = (_Float16)vis;
    float v2;
    if (lane == 0) {
      float cpx = center[0], cpy = center[1];
      float pcx = px * 8.0f + 4.0f, pcy = py * 8.0f + 4.0f;
      float dist = fmaxf(fabsf(pcx - cpx), fabsf(pcy - cpy));
      float md = fmaxf(fmaxf(fabsf(cpx), fabsf(cpy)),
                       fmaxf(fabsf(cpx - 1024.0f), fabsf(cpy - 1024.0f)));
      v2 = dist / md;                          // nd at col 32
    } else {
      v2 = s_textf[lane - 1];                  // text cols 33..63
    }
    s_comb[w][32 + lane] = (_Float16)v2;
    s_comb[w][64 + lane] = (_Float16)((lane == 0) ? s_textf[31] : 0.0f);
  }
  __syncthreads();

  // ====== phase 5: MLP 65->64 relu (wave 0; K pad 96, N=64: 12 WMMA) ======
  if (w == 0) {
    v8f h0 = {}, h1v = {}, h2v = {}, h3 = {};
#pragma unroll
    for (int ks = 0; ks < 3; ks++) {
      v16h af;
      v8h* ap = (v8h*)&af;
      ap[0] = *reinterpret_cast<const v8h*>(&s_comb[lo][32 * ks + 8 * hi]);
      ap[1] = *reinterpret_cast<const v8h*>(&s_comb[lo][32 * ks + 16 + 8 * hi]);
#pragma unroll
      for (int nt = 0; nt < 4; nt++) {
        v16h bf;
        int n = nt * 16 + lo;
#pragma unroll
        for (int e = 0; e < 16; e++) {
          int k = 32 * ks + e + 8 * (e >> 3) + 8 * hi;
          bf[e] = (k < 65) ? (_Float16)m1w[k * 64 + n] : (_Float16)0.0f;
        }
        v8f* acc = (nt == 0) ? &h0 : (nt == 1) ? &h1v : (nt == 2) ? &h2v : &h3;
        *acc = __builtin_amdgcn_wmma_f32_16x16x32_f16(false, af, false, bf,
                                                      (short)0, *acc, false, false);
      }
    }
#pragma unroll
    for (int nt = 0; nt < 4; nt++) {
      v8f acc = (nt == 0) ? h0 : (nt == 1) ? h1v : (nt == 2) ? h2v : h3;
      int n = nt * 16 + lo;
      float bias = m1b[n];
#pragma unroll
      for (int r = 0; r < 8; r++)
        s_h1[r + 8 * hi][n] = (_Float16)fmaxf(acc[r] + bias, 0.0f);
    }
  }
  __syncthreads();

  // ====== phase 6: MLP 64->32 relu (4 WMMA) ======
  if (w == 0) {
    v8f g0 = {}, g1 = {};
#pragma unroll
    for (int ks = 0; ks < 2; ks++) {
      v16h af;
      v8h* ap = (v8h*)&af;
      ap[0] = *reinterpret_cast<const v8h*>(&s_h1[lo][32 * ks + 8 * hi]);
      ap[1] = *reinterpret_cast<const v8h*>(&s_h1[lo][32 * ks + 16 + 8 * hi]);
#pragma unroll
      for (int nt = 0; nt < 2; nt++) {
        v16h bf;
        int n = nt * 16 + lo;
#pragma unroll
        for (int e = 0; e < 16; e++) bf[e] = (_Float16)m2w[(32 * ks + e + 8 * (e >> 3) + 8 * hi) * 32 + n];
        if (nt == 0)
          g0 = __builtin_amdgcn_wmma_f32_16x16x32_f16(false, af, false, bf,
                                                      (short)0, g0, false, false);
        else
          g1 = __builtin_amdgcn_wmma_f32_16x16x32_f16(false, af, false, bf,
                                                      (short)0, g1, false, false);
      }
    }
#pragma unroll
    for (int nt = 0; nt < 2; nt++) {
      v8f acc = (nt == 0) ? g0 : g1;
      int n = nt * 16 + lo;
      float bias = m2b[n];
#pragma unroll
      for (int r = 0; r < 8; r++)
        s_h2[r + 8 * hi][n] = (_Float16)fmaxf(acc[r] + bias, 0.0f);
    }
  }
  __syncthreads();

  // ====== phase 7: MLP 32->1 (1 WMMA, B padded to N=16) + sigmoid/clamp ======
  if (w == 0) {
    v16h af, bf;
    v8h* ap = (v8h*)&af;
    ap[0] = *reinterpret_cast<const v8h*>(&s_h2[lo][8 * hi]);
    ap[1] = *reinterpret_cast<const v8h*>(&s_h2[lo][16 + 8 * hi]);
#pragma unroll
    for (int e = 0; e < 16; e++)
      bf[e] = (lo == 0) ? (_Float16)m3w[e + 8 * (e >> 3) + 8 * hi] : (_Float16)0.0f;
    v8f d = {};
    d = __builtin_amdgcn_wmma_f32_16x16x32_f16(false, af, false, bf,
                                               (short)0, d, false, false);
    if (lo == 0) {   // lanes 0 and 16 hold column N=0 (rows r+8*hi)
      float bias = m3b[0];
#pragma unroll
      for (int r = 0; r < 8; r++) {
        int m = r + 8 * hi;
        float s = d[r] + bias;
        float sig = 1.0f / (1.0f + __expf(-s));
        float sp = fminf(fmaxf(5.0f * sig + 3.0f, 3.0f), 15.0f);
        sparsity_out[(size_t)blockIdx.x * WAVES + m] = sp;
      }
    }
  }
}

extern "C" void kernel_launch(void* const* d_in, const int* in_sizes, int n_in,
                              void* d_out, int out_size, void* d_ws, size_t ws_size,
                              hipStream_t stream) {
  (void)in_sizes; (void)n_in; (void)out_size; (void)d_ws; (void)ws_size;
  const float* x   = (const float*)d_in[0];
  const float* te  = (const float*)d_in[1];
  // d_in[2] = mask: unused (update term is exactly zero -> x_out == x)
  const float* cp  = (const float*)d_in[3];
  const float* trw = (const float*)d_in[4];
  const float* trb = (const float*)d_in[5];
  const float* c1w = (const float*)d_in[6];
  const float* c1b = (const float*)d_in[7];
  const float* c2w = (const float*)d_in[8];
  const float* c2b = (const float*)d_in[9];
  const float* m1w = (const float*)d_in[10];
  const float* m1b = (const float*)d_in[11];
  const float* m2w = (const float*)d_in[12];
  const float* m2b = (const float*)d_in[13];
  const float* m3w = (const float*)d_in[14];
  const float* m3b = (const float*)d_in[15];

  const size_t x_elems = (size_t)4 * 1024 * 1024;   // B*C*H*W
  float* out = (float*)d_out;

  // x_out = x (identity write-back); memory-bound: 32 MB @ 23.3 TB/s ~ 1.4 us
  hipMemcpyAsync(out, x, x_elems * sizeof(float), hipMemcpyDeviceToDevice, stream);

  // 65536 patches / 16 per workgroup
  fused_patch_kernel<<<4096, 32 * WAVES, 0, stream>>>(
      x, te, cp, trw, trb, c1w, c1b, c2w, c2b,
      m1w, m1b, m2w, m2b, m3w, m3b, out + x_elems);
}
